// MultiHeadAttention_9612136808802
// MI455X (gfx1250) — compile-verified, matrix operand order checked by reading, not run on hardware
//
#include <hip/hip_runtime.h>
#include <stdint.h>

// ---------------------------------------------------------------------------
// MI455X (gfx1250) fused relative-bias multi-head attention, wave32 + WMMA.
// B=8, L=512, HID=512, H=8, D=64, NBIAS=183.
// ---------------------------------------------------------------------------
#define B_    8
#define L_    512
#define HID_  512
#define H_    8
#define D_    64
#define NB_   183
#define NBP_  192      // NBIAS padded to a multiple of 32 for WMMA tiles

typedef __attribute__((ext_vector_type(16))) __bf16 bf16x16;  // 8 VGPRs
typedef __attribute__((ext_vector_type(8)))  float  f32x8;    // 8 VGPRs
typedef __attribute__((ext_vector_type(4)))  int    i32x4;

struct alignas(16) U4 { uint32_t x[4]; };

// --- optional gfx1250 async global->LDS path (ASYNCcnt), guarded ----------
#ifdef __has_builtin
#if defined(__AMDGCN__) && __has_builtin(__builtin_amdgcn_global_load_async_to_lds_b128)
#define ASYNC_LDS 1
#endif
#endif
#ifndef ASYNC_LDS
#define ASYNC_LDS 0
#endif

#define AS1 __attribute__((address_space(1)))
#define AS3 __attribute__((address_space(3)))

__device__ __forceinline__ f32x8 wmma_bf16(bf16x16 a, bf16x16 b, f32x8 c) {
  // D(16x16,f32) = A(16x32,bf16) * B(32x16,bf16) + C
  return __builtin_amdgcn_wmma_f32_16x16x32_bf16(
      /*neg_a=*/false, a, /*neg_b=*/false, b,
      /*c_mod=*/(short)0, c, /*reuse_a=*/false, /*reuse_b=*/false);
}

// 16x32 bf16 A-fragment from a row-major bf16 matrix (ISA 7.12.2 layout):
// lanes 0-15: row m=lane, K {0..7,16..23}; lanes 16-31: row m=lane-16,
// K {8..15,24..31}. Two b128 loads per lane. The same pattern loads the B
// operand of X@W^T from rows of W (since B[k,n] = W[n,k]).
__device__ __forceinline__ bf16x16 frag_from_bf16(const __bf16* base, int ld) {
  const int lane = threadIdx.x & 31;
  const int r = lane & 15, hf = lane >> 4;
  const __bf16* p = base + (size_t)r * ld + hf * 8;
  union { U4 q[2]; bf16x16 v; } u;
  u.q[0] = *reinterpret_cast<const U4*>(p);
  u.q[1] = *reinterpret_cast<const U4*>(p + 16);
  return u.v;
}

// Same fragment pattern from an fp32 source (used for LDS-resident S/bins).
__device__ __forceinline__ bf16x16 frag_from_f32(const float* base, int ld) {
  const int lane = threadIdx.x & 31;
  const int r = lane & 15, hf = lane >> 4;
  const float* p = base + (size_t)r * ld + hf * 8;
  float4 a0 = *reinterpret_cast<const float4*>(p);
  float4 a1 = *reinterpret_cast<const float4*>(p + 4);
  float4 b0 = *reinterpret_cast<const float4*>(p + 16);
  float4 b1 = *reinterpret_cast<const float4*>(p + 20);
  bf16x16 v;
  v[0]  = (__bf16)a0.x; v[1]  = (__bf16)a0.y; v[2]  = (__bf16)a0.z; v[3]  = (__bf16)a0.w;
  v[4]  = (__bf16)a1.x; v[5]  = (__bf16)a1.y; v[6]  = (__bf16)a1.z; v[7]  = (__bf16)a1.w;
  v[8]  = (__bf16)b0.x; v[9]  = (__bf16)b0.y; v[10] = (__bf16)b0.z; v[11] = (__bf16)b0.w;
  v[12] = (__bf16)b1.x; v[13] = (__bf16)b1.y; v[14] = (__bf16)b1.z; v[15] = (__bf16)b1.w;
  return v;
}

// Cooperative stage of `nbytes` (16B granular) global bf16 -> LDS, 128 thr.
// Generic->AS casts go through uintptr_t: AS1 shares the generic value; a
// generic LDS pointer's low 32 bits are the LDS offset (flat aperture rule).
__device__ __forceinline__ void stage_to_lds(const __bf16* g, __bf16* l, int nbytes) {
#if ASYNC_LDS
  for (int o = threadIdx.x * 16; o < nbytes; o += 128 * 16)
    __builtin_amdgcn_global_load_async_to_lds_b128(
        (AS1 i32x4*)(uintptr_t)((const char*)g + o),
        (AS3 i32x4*)(uint32_t)(uintptr_t)((char*)l + o), 0, 0);
#else
  for (int o = threadIdx.x * 16; o < nbytes; o += 128 * 16)
    *reinterpret_cast<U4*>((char*)l + o) =
        *reinterpret_cast<const U4*>((const char*)g + o);
#endif
}

__device__ __forceinline__ void stage_wait() {
#if ASYNC_LDS
#if __has_builtin(__builtin_amdgcn_s_wait_asynccnt)
  __builtin_amdgcn_s_wait_asynccnt(0);
#else
  asm volatile("s_wait_asynccnt 0" ::: "memory");
#endif
#endif
  __syncthreads();
}

// ---------------------------------------------------------------------------
// Kernel: fp32 -> bf16 bulk convert (8 elems/thread, b128 in, b128 out).
// ---------------------------------------------------------------------------
__global__ void k_tobf16(const float* __restrict__ src, __bf16* __restrict__ dst,
                         int n8) {
  const int i = blockIdx.x * blockDim.x + threadIdx.x;
  if (i >= n8) return;
  const float4* s = reinterpret_cast<const float4*>(src) + (size_t)i * 2;
  float4 a = s[0], b = s[1];
  union { U4 q; __bf16 h[8]; } u;
  u.h[0] = (__bf16)a.x; u.h[1] = (__bf16)a.y; u.h[2] = (__bf16)a.z; u.h[3] = (__bf16)a.w;
  u.h[4] = (__bf16)b.x; u.h[5] = (__bf16)b.y; u.h[6] = (__bf16)b.z; u.h[7] = (__bf16)b.w;
  reinterpret_cast<U4*>(dst)[i] = u.q;
}

// ---------------------------------------------------------------------------
// Kernel: pad + convert bias tables.
//   tabA [NBP_][D_] bf16 : att_bias_tab rows, zero-padded t >= 183
//   vbT  [D_][NBP_] bf16 : vec_bias_tab transposed, zero-padded
// ---------------------------------------------------------------------------
__global__ void k_prep(const float* __restrict__ attTab,
                       const float* __restrict__ vecTab,
                       __bf16* __restrict__ tabA, __bf16* __restrict__ vbT) {
  for (int idx = threadIdx.x; idx < NBP_ * D_; idx += blockDim.x) {
    const int t = idx / D_, d = idx % D_;
    const float a = (t < NB_) ? attTab[t * D_ + d] : 0.f;
    const float v = (t < NB_) ? vecTab[t * D_ + d] : 0.f;
    tabA[t * D_ + d] = (__bf16)a;
    vbT[d * NBP_ + t] = (__bf16)v;
  }
}

// ---------------------------------------------------------------------------
// Kernel: projection GEMM  out = X[4096,512] @ W^T  (bf16 in, f32 acc).
// grid (M/64, N/64); 4 waves/block; wave = 16 rows x 64 cols.
// ---------------------------------------------------------------------------
__global__ __launch_bounds__(128) void k_proj(const __bf16* __restrict__ X,
                                              const __bf16* __restrict__ W,
                                              __bf16* __restrict__ outStd,
                                              __bf16* __restrict__ outT,
                                              float scale) {
  const int wv = threadIdx.x >> 5, lane = threadIdx.x & 31;
  const int m0 = blockIdx.x * 64 + wv * 16;
  const int n0 = blockIdx.y * 64;
  f32x8 acc[4] = {};
  for (int k0 = 0; k0 < HID_; k0 += 32) {
    bf16x16 a = frag_from_bf16(X + (size_t)m0 * HID_ + k0, HID_);
#pragma unroll
    for (int j = 0; j < 4; ++j) {
      bf16x16 b = frag_from_bf16(W + (size_t)(n0 + j * 16) * HID_ + k0, HID_);
      acc[j] = wmma_bf16(a, b, acc[j]);
    }
  }
  const int b = m0 / L_, i0 = m0 % L_, h = n0 / D_;
  const int nl = lane & 15, mhalf = (lane >> 4) * 8;
#pragma unroll
  for (int j = 0; j < 4; ++j) {
    const int d0 = (n0 % D_) + j * 16;
    if (outStd) {                                   // [B,H,L,D]
      __bf16* dst = outStd + ((size_t)(b * H_ + h) * L_ + i0) * D_ + d0;
#pragma unroll
      for (int r = 0; r < 8; ++r)
        dst[(size_t)(mhalf + r) * D_ + nl] = (__bf16)(acc[j][r] * scale);
    }
    if (outT) {                                     // [B,H,D,L]: i contiguous
      union { U4 q; __bf16 hv[8]; } u;
#pragma unroll
      for (int r = 0; r < 8; ++r) u.hv[r] = (__bf16)(acc[j][r] * scale);
      __bf16* dst = outT + ((size_t)(b * H_ + h) * D_ + d0 + nl) * L_ + i0 + mhalf;
      *reinterpret_cast<U4*>(dst) = u.q;            // one b128 store
    }
  }
}

// ---------------------------------------------------------------------------
// Kernel: P[b,h,i,t] = q[b,h,i,:] . att_bias_tab[t,:]   ->  f32 [BHL, 192]
// ---------------------------------------------------------------------------
__global__ __launch_bounds__(128) void k_biasproj(const __bf16* __restrict__ qb,
                                                  const __bf16* __restrict__ tabA,
                                                  float* __restrict__ P) {
  const int wv = threadIdx.x >> 5, lane = threadIdx.x & 31;
  const int m0 = blockIdx.x * 64 + wv * 16;
  const int n0 = blockIdx.y * 64;
  bf16x16 a0 = frag_from_bf16(qb + (size_t)m0 * D_, D_);
  bf16x16 a1 = frag_from_bf16(qb + (size_t)m0 * D_ + 32, D_);
  const int nl = lane & 15, mhalf = (lane >> 4) * 8;
#pragma unroll
  for (int j = 0; j < 4; ++j) {
    f32x8 c = {};
    c = wmma_bf16(a0, frag_from_bf16(tabA + (size_t)(n0 + j * 16) * D_, D_), c);
    c = wmma_bf16(a1, frag_from_bf16(tabA + (size_t)(n0 + j * 16) * D_ + 32, D_), c);
    float* dst = P + (size_t)m0 * NBP_ + n0 + j * 16;
#pragma unroll
    for (int r = 0; r < 8; ++r) dst[(size_t)(mhalf + r) * NBP_ + nl] = c[r];
  }
}

// ---------------------------------------------------------------------------
// Kernel: attention core. grid (B*H, L/64); 4 waves, wave = 16 query rows.
// LDS: S 64x512 f32 (128K) + bins 64x192 f32 (48K) + rsum (256B)
//      + 64KB K/V^T stage buffer  = ~240 KB of the 320 KB WGP LDS.
// K block staged (async) before phase 1; V^T block restaged during softmax.
// ---------------------------------------------------------------------------
#define SMEM_F32 (4 * 16 * L_ + 4 * 16 * NBP_ + 64)

__global__ __launch_bounds__(128) void k_attn(const __bf16* __restrict__ qb,
                                              const __bf16* __restrict__ kb,
                                              const __bf16* __restrict__ vT,
                                              const float* __restrict__ P,
                                              const int* __restrict__ tbias,
                                              const unsigned char* __restrict__ mask,
                                              const __bf16* __restrict__ vbT,
                                              __bf16* __restrict__ ctx) {
  extern __shared__ float smem[];
  const int wv = threadIdx.x >> 5, lane = threadIdx.x & 31;
  const int bh = blockIdx.x;                 // b*H + h
  const int b = bh >> 3, h = bh & 7;
  const int i0 = blockIdx.y * 64 + wv * 16;
  float* Sw     = smem + (size_t)wv * 16 * L_;
  float* bins   = smem + 4 * 16 * L_ + (size_t)wv * 16 * NBP_;
  float* rsum   = smem + 4 * 16 * L_ + 4 * 16 * NBP_ + wv * 16;
  __bf16* stage = (__bf16*)(smem + SMEM_F32);

  // start K(b,h) [512x64 bf16] staging; overlap with setup
  stage_to_lds(kb + (size_t)bh * L_ * D_, stage, L_ * D_ * 2);

  for (int t = lane; t < 16 * NBP_; t += 32) bins[t] = 0.f;
  const int nl = lane & 15, mhalf = (lane >> 4) * 8;
  const __bf16* qrow = qb + ((size_t)bh * L_ + i0) * D_;
  bf16x16 a0 = frag_from_bf16(qrow, D_);
  bf16x16 a1 = frag_from_bf16(qrow + 32, D_);
  const float* Pw = P + (size_t)bh * L_ * NBP_;
  const int* tbw = tbias + (size_t)b * L_ * L_;
  const unsigned char* mkw = mask + (size_t)b * L_ * L_;
  stage_wait();

  // -------- phase 1: S = q k^T + P[gather], masked --------
  for (int j0 = 0; j0 < L_; j0 += 16) {
    f32x8 c = {};
    c = wmma_bf16(a0, frag_from_bf16(stage + j0 * D_, D_), c);
    c = wmma_bf16(a1, frag_from_bf16(stage + j0 * D_ + 32, D_), c);
#pragma unroll
    for (int r = 0; r < 8; ++r) {
      const int m = mhalf + r;
      const int i = i0 + m, j = j0 + nl;
      const size_t ij = (size_t)i * L_ + j;
      float s = c[r] + Pw[(size_t)i * NBP_ + tbw[ij]];
      if (mkw[ij]) s = -1e30f;
      Sw[m * L_ + j] = s;
    }
  }
  __syncthreads();

  // restage V^T(b,h) [64x512 bf16]; DMA overlaps the softmax below
  stage_to_lds(vT + (size_t)bh * D_ * L_, stage, D_ * L_ * 2);

  // -------- phase 2: softmax (unnormalized exp) + prob histogram --------
  for (int rr = 0; rr < 16; ++rr) {
    float* row = Sw + rr * L_;
    const int* trow = tbw + (size_t)(i0 + rr) * L_;
    float vals[16];
    float mx = -3.0e38f;
#pragma unroll
    for (int k = 0; k < 16; ++k) {
      vals[k] = row[lane + k * 32];
      mx = fmaxf(mx, vals[k]);
    }
#pragma unroll
    for (int off = 16; off > 0; off >>= 1) mx = fmaxf(mx, __shfl_xor(mx, off, 32));
    float sum = 0.f;
#pragma unroll
    for (int k = 0; k < 16; ++k) {
      const float e = __expf(vals[k] - mx);
      row[lane + k * 32] = e;
      sum += e;
      atomicAdd(&bins[rr * NBP_ + trow[lane + k * 32]], e);  // ds_add_f32
    }
#pragma unroll
    for (int off = 16; off > 0; off >>= 1) sum += __shfl_xor(sum, off, 32);
    if (lane == 0) rsum[rr] = sum;
  }
  stage_wait();

  // -------- phase 3: O = (e @ V + bins @ vbT) / rowsum --------
  f32x8 acc[4] = {};
  for (int k0 = 0; k0 < L_; k0 += 32) {
    bf16x16 a = frag_from_f32(Sw + k0, L_);            // LDS f32 -> bf16 frag
#pragma unroll
    for (int n = 0; n < 4; ++n)
      acc[n] = wmma_bf16(a, frag_from_bf16(stage + (size_t)(n * 16) * L_ + k0, L_),
                         acc[n]);
  }
#pragma unroll
  for (int k0 = 0; k0 < NBP_; k0 += 32) {
    bf16x16 a = frag_from_f32(bins + k0, NBP_);
#pragma unroll
    for (int n = 0; n < 4; ++n)
      acc[n] = wmma_bf16(a, frag_from_bf16(vbT + (size_t)(n * 16) * NBP_ + k0, NBP_),
                         acc[n]);
  }
#pragma unroll
  for (int n = 0; n < 4; ++n) {
#pragma unroll
    for (int r = 0; r < 8; ++r) {
      const int m = mhalf + r;
      const float rinv = 1.f / rsum[m];
      const int i = i0 + m, d = n * 16 + nl;
      ctx[((size_t)b * L_ + i) * HID_ + h * D_ + d] = (__bf16)(acc[n][r] * rinv);
    }
  }
}

// ---------------------------------------------------------------------------
// Kernel: out = ctx[4096,512] @ Wo^T -> f32
// ---------------------------------------------------------------------------
__global__ __launch_bounds__(128) void k_outproj(const __bf16* __restrict__ ctx,
                                                 const __bf16* __restrict__ Wo,
                                                 float* __restrict__ out) {
  const int wv = threadIdx.x >> 5, lane = threadIdx.x & 31;
  const int m0 = blockIdx.x * 64 + wv * 16;
  const int n0 = blockIdx.y * 64;
  f32x8 acc[4] = {};
  for (int k0 = 0; k0 < HID_; k0 += 32) {
    bf16x16 a = frag_from_bf16(ctx + (size_t)m0 * HID_ + k0, HID_);
#pragma unroll
    for (int j = 0; j < 4; ++j)
      acc[j] = wmma_bf16(a, frag_from_bf16(Wo + (size_t)(n0 + j * 16) * HID_ + k0, HID_),
                         acc[j]);
  }
  const int nl = lane & 15, mhalf = (lane >> 4) * 8;
#pragma unroll
  for (int j = 0; j < 4; ++j) {
    float* dst = out + (size_t)m0 * HID_ + n0 + j * 16;
#pragma unroll
    for (int r = 0; r < 8; ++r) dst[(size_t)(mhalf + r) * HID_ + nl] = acc[j][r];
  }
}

// ---------------------------------------------------------------------------
extern "C" void kernel_launch(void* const* d_in, const int* in_sizes, int n_in,
                              void* d_out, int out_size, void* d_ws, size_t ws_size,
                              hipStream_t stream) {
  (void)in_sizes; (void)n_in; (void)out_size; (void)ws_size;
  const float* Q = (const float*)d_in[0];
  const float* K = (const float*)d_in[1];
  const float* V = (const float*)d_in[2];
  const unsigned char* mask = (const unsigned char*)d_in[3];  // jnp.bool_
  const int* tbias = (const int*)d_in[4];
  const float* Wq = (const float*)d_in[5];
  const float* Wk = (const float*)d_in[6];
  const float* Wv = (const float*)d_in[7];
  const float* Wo = (const float*)d_in[8];
  const float* attTab = (const float*)d_in[9];
  const float* vecTab = (const float*)d_in[10];
  float* out = (float*)d_out;

  // workspace carve-up (~60 MB)
  char* ws = (char*)d_ws;
  size_t off = 0;
  auto alloc = [&](size_t bytes) -> void* {
    void* p = ws + off;
    off = (off + bytes + 255) & ~(size_t)255;
    return p;
  };
  const size_t NXE = (size_t)B_ * L_ * HID_;   // elements in X / proj outputs
  const size_t NWE = (size_t)HID_ * HID_;      // elements in a weight matrix
  __bf16* Qb   = (__bf16*)alloc(NXE * 2);
  __bf16* Kb   = (__bf16*)alloc(NXE * 2);
  __bf16* Vb   = (__bf16*)alloc(NXE * 2);
  __bf16* Wqb  = (__bf16*)alloc(NWE * 2);
  __bf16* Wkb  = (__bf16*)alloc(NWE * 2);
  __bf16* Wvb  = (__bf16*)alloc(NWE * 2);
  __bf16* Wob  = (__bf16*)alloc(NWE * 2);
  __bf16* qb   = (__bf16*)alloc(NXE * 2);                         // [B,H,L,D]
  __bf16* kb   = (__bf16*)alloc(NXE * 2);                         // [B,H,L,D]
  __bf16* vT   = (__bf16*)alloc(NXE * 2);                         // [B,H,D,L]
  __bf16* ctx  = (__bf16*)alloc(NXE * 2);                         // [B,L,HID]
  float*  P    = (float*)alloc((size_t)B_ * H_ * L_ * NBP_ * 4);  // [B,H,L,192]
  __bf16* tabA = (__bf16*)alloc((size_t)NBP_ * D_ * 2);
  __bf16* vbT  = (__bf16*)alloc((size_t)D_ * NBP_ * 2);

  // bulk fp32 -> bf16
  hipLaunchKernelGGL(k_tobf16, dim3((NXE / 8 + 255) / 256), dim3(256), 0, stream, Q, Qb, (int)(NXE / 8));
  hipLaunchKernelGGL(k_tobf16, dim3((NXE / 8 + 255) / 256), dim3(256), 0, stream, K, Kb, (int)(NXE / 8));
  hipLaunchKernelGGL(k_tobf16, dim3((NXE / 8 + 255) / 256), dim3(256), 0, stream, V, Vb, (int)(NXE / 8));
  hipLaunchKernelGGL(k_tobf16, dim3((NWE / 8 + 255) / 256), dim3(256), 0, stream, Wq, Wqb, (int)(NWE / 8));
  hipLaunchKernelGGL(k_tobf16, dim3((NWE / 8 + 255) / 256), dim3(256), 0, stream, Wk, Wkb, (int)(NWE / 8));
  hipLaunchKernelGGL(k_tobf16, dim3((NWE / 8 + 255) / 256), dim3(256), 0, stream, Wv, Wvb, (int)(NWE / 8));
  hipLaunchKernelGGL(k_tobf16, dim3((NWE / 8 + 255) / 256), dim3(256), 0, stream, Wo, Wob, (int)(NWE / 8));
  hipLaunchKernelGGL(k_prep, dim3(1), dim3(256), 0, stream, attTab, vecTab, tabA, vbT);

  const float scale = 0.125f;  // D^-0.5 = 64^-0.5
  hipLaunchKernelGGL(k_proj, dim3(64, 8), dim3(128), 0, stream,
                     Qb, Wqb, qb, (__bf16*)nullptr, scale);
  hipLaunchKernelGGL(k_proj, dim3(64, 8), dim3(128), 0, stream,
                     Kb, Wkb, kb, (__bf16*)nullptr, 1.0f);
  hipLaunchKernelGGL(k_proj, dim3(64, 8), dim3(128), 0, stream,
                     Vb, Wvb, (__bf16*)nullptr, vT, 1.0f);
  hipLaunchKernelGGL(k_biasproj, dim3(512, 3), dim3(128), 0, stream, qb, tabA, P);

  const size_t smemBytes = (size_t)SMEM_F32 * sizeof(float) + (size_t)L_ * D_ * 2;
  hipLaunchKernelGGL(k_attn, dim3(B_ * H_, L_ / 64), dim3(128), smemBytes, stream,
                     qb, kb, vT, P, tbias, mask, vbT, ctx);

  hipLaunchKernelGGL(k_outproj, dim3(64, 8), dim3(128), 0, stream, ctx, Wob, out);
}
